// TransformerEncoderLayer_5677946765670
// MI455X (gfx1250) — matrix operand order repaired and checked by reading, not compiled, not run on hardware
//
#include <hip/hip_runtime.h>
#include <hip/hip_bf16.h>

typedef __attribute__((ext_vector_type(16))) __bf16 v16bf;
typedef __attribute__((ext_vector_type(8)))  float  v8f;

// ---------- bf16 helpers ----------
__device__ __forceinline__ unsigned short f2bf(float f) {
  unsigned int u = __builtin_bit_cast(unsigned int, f);
  u += 0x7FFFu + ((u >> 16) & 1u);
  return (unsigned short)(u >> 16);
}
// Pack two floats into two bf16 (one v_cvt_pk_bf16_f32 when available).
__device__ __forceinline__ unsigned int pack2bf(float lo, float hi) {
#if __has_builtin(__builtin_amdgcn_cvt_pk_bf16_f32)
  typedef __attribute__((ext_vector_type(2))) __bf16 v2bf;
  v2bf v = __builtin_amdgcn_cvt_pk_bf16_f32(lo, hi);
  return __builtin_bit_cast(unsigned int, v);
#else
  return (unsigned)f2bf(lo) | ((unsigned)f2bf(hi) << 16);
#endif
}

struct u4x2 { uint4 a, b; };  // 32 bytes == sizeof(v16bf)

// ---------- WMMA fragment loaders: 2 x ds_load_b128, zero packing ----------
// A fragment: 16(M) x 32(K) bf16, LDS row-major [m][k], element stride `stride`.
__device__ __forceinline__ v16bf load_frag_a(const unsigned short* p, int stride) {
  const int lane = threadIdx.x & 31;
  const unsigned short* q = p + (lane & 15) * stride + ((lane < 16) ? 0 : 8);
  u4x2 t;
  t.a = *(const uint4*)q;         // K = kb+0..7
  t.b = *(const uint4*)(q + 16);  // K = kb+16..23
  return __builtin_bit_cast(v16bf, t);
}
// B fragment: 32(K) x 16(N) bf16, LDS stored N-MAJOR [n][k] (contiguous in K).
__device__ __forceinline__ v16bf load_frag_b_nm(const unsigned short* p, int stride) {
  const int lane = threadIdx.x & 31;
  const unsigned short* q = p + (lane & 15) * stride + ((lane < 16) ? 0 : 8);
  u4x2 t;
  t.a = *(const uint4*)q;
  t.b = *(const uint4*)(q + 16);
  return __builtin_bit_cast(v16bf, t);
}
// Fragment-linear A operand (each lane's 16 elements stored contiguously).
__device__ __forceinline__ v16bf load_frag_lin(const unsigned short* p) {
  const unsigned short* q = p + (threadIdx.x & 31) * 16;
  u4x2 t;
  t.a = *(const uint4*)q;
  t.b = *(const uint4*)(q + 8);
  return __builtin_bit_cast(v16bf, t);
}
__device__ __forceinline__ v8f wmma_bf16(v16bf a, v16bf b, v8f c) {
  return __builtin_amdgcn_wmma_f32_16x16x32_bf16(false, a, false, b, (short)0, c,
                                                 false, false);
}

// ---------- CDNA5 async LDS fill (GLOBAL_LOAD_ASYNC_TO_LDS_B128) ----------
// Per-lane: copies 16 bytes from g_src into LDS at lds_dst; tracked by ASYNCcnt.
__device__ __forceinline__ void async_b128(const unsigned short* lds_dst,
                                           const unsigned short* g_src) {
  const unsigned off = (unsigned)(uintptr_t)lds_dst;
  asm volatile("global_load_async_to_lds_b128 %0, %1, off"
               :: "v"(off), "v"(g_src) : "memory");
}
__device__ __forceinline__ void wait_async0() {
  asm volatile("s_wait_asynccnt 0" ::: "memory");
}

// =============================================================================
// Tiled GEMM: Out[M,N] = act( A[M,K] @ (W[K,N]*wscale) + bias*wscale [+resid] )
// BM=128 BN=128 BK=64, 256 thr = 8 waves, wave -> 64x32 (4x2 frags).
// =============================================================================
template <bool A_BF16, bool OUT_BF16, bool RELU, bool RESID>
__global__ __launch_bounds__(256) void gemm_kernel(
    const void* __restrict__ Aop, const float* __restrict__ W,
    const float* __restrict__ bias, const float* __restrict__ resid,
    void* __restrict__ Oop, int M, int N, int K, float wscale) {
  constexpr int BM = 128, BN = 128, BK = 64;
  constexpr int AS  = BK + 8;  // 72: A tile [m][k]
  constexpr int BSK = BK + 8;  // 72: W tile N-major [n][k]
  __shared__ unsigned short sA[BM * AS];
  __shared__ unsigned short sBT[BN * BSK];

  const int tid = threadIdx.x, lane = tid & 31, wave = tid >> 5;
  const int bn = blockIdx.x, bm = blockIdx.y;
  const int wm = wave & 1;
  const int wn = wave >> 1;
  const int nloc  = lane & 15;
  const int rhalf = (lane >> 4) << 3;

  v8f acc[4][2];
#pragma unroll
  for (int j = 0; j < 2; ++j) {
    const float bv = bias[bn * BN + wn * 32 + j * 16 + nloc] * wscale;
#pragma unroll
    for (int i = 0; i < 4; ++i)
#pragma unroll
      for (int r = 0; r < 8; ++r) acc[i][j][r] = bv;
  }

  for (int kb = 0; kb < K; kb += BK) {
    if (kb + BK < K) {  // prefetch next stage (global_prefetch_b8)
      __builtin_prefetch(W + (size_t)(kb + BK + (tid >> 2)) * N + bn * BN + (tid & 3) * 32, 0, 1);
      if constexpr (A_BF16)
        __builtin_prefetch((const unsigned short*)Aop + (size_t)(bm * BM + (tid >> 1)) * K + kb + BK + (tid & 1) * 32, 0, 1);
      else
        __builtin_prefetch((const float*)Aop + (size_t)(bm * BM + (tid >> 1)) * K + kb + BK + (tid & 1) * 32, 0, 1);
    }
    // ---- stage A tile (128 x 64) row-major bf16 ----
    if constexpr (A_BF16) {
      const unsigned short* A = (const unsigned short*)Aop;
#pragma unroll
      for (int i = 0; i < 4; ++i) {
        const int idx = tid + i * 256;  // 1024 x uint4 (8 bf16)
        const int row = idx >> 3, c8 = idx & 7;
        *(uint4*)&sA[row * AS + c8 * 8] =
            *(const uint4*)(A + (size_t)(bm * BM + row) * K + kb + c8 * 8);
      }
    } else {
      const float* A = (const float*)Aop;
#pragma unroll
      for (int i = 0; i < 8; ++i) {
        const int idx = tid + i * 256;  // 2048 x float4
        const int row = idx >> 4, c4 = idx & 15;
        const float4 f = *(const float4*)(A + (size_t)(bm * BM + row) * K + kb + c4 * 4);
        uint2 pk;
        pk.x = pack2bf(f.x, f.y);
        pk.y = pack2bf(f.z, f.w);
        *(uint2*)&sA[row * AS + c4 * 4] = pk;
      }
    }
    // ---- stage W tile (64 x 128) transposed to N-major [n][k]:
    //      two k-rows per thread -> packed ds_store_b32 along k ----
#pragma unroll
    for (int i = 0; i < 4; ++i) {
      const int idx = tid + i * 256;   // 1024 tasks: 32 row-pairs x 32 col4
      const int k0 = (idx >> 5) * 2;   // even k
      const int c4 = idx & 31;
      const float4 f0 = *(const float4*)(W + (size_t)(kb + k0) * N + bn * BN + c4 * 4);
      const float4 f1 = *(const float4*)(W + (size_t)(kb + k0 + 1) * N + bn * BN + c4 * 4);
      const int n0 = c4 * 4;
      *(unsigned*)&sBT[(n0 + 0) * BSK + k0] = pack2bf(f0.x * wscale, f1.x * wscale);
      *(unsigned*)&sBT[(n0 + 1) * BSK + k0] = pack2bf(f0.y * wscale, f1.y * wscale);
      *(unsigned*)&sBT[(n0 + 2) * BSK + k0] = pack2bf(f0.z * wscale, f1.z * wscale);
      *(unsigned*)&sBT[(n0 + 3) * BSK + k0] = pack2bf(f0.w * wscale, f1.w * wscale);
    }
    __syncthreads();

#pragma unroll
    for (int ks = 0; ks < 2; ++ks) {
      v16bf afr[4];
#pragma unroll
      for (int i = 0; i < 4; ++i)
        afr[i] = load_frag_a(&sA[(wm * 64 + i * 16) * AS + ks * 32], AS);
      v16bf bfr[2];
#pragma unroll
      for (int j = 0; j < 2; ++j)
        bfr[j] = load_frag_b_nm(&sBT[(wn * 32 + j * 16) * BSK + ks * 32], BSK);
#pragma unroll
      for (int i = 0; i < 4; ++i)
#pragma unroll
        for (int j = 0; j < 2; ++j) acc[i][j] = wmma_bf16(afr[i], bfr[j], acc[i][j]);
    }
    __syncthreads();
  }

  // ---- epilogue ----
#pragma unroll
  for (int i = 0; i < 4; ++i)
#pragma unroll
    for (int j = 0; j < 2; ++j) {
      const int n = bn * BN + wn * 32 + j * 16 + nloc;
#pragma unroll
      for (int r = 0; r < 8; ++r) {
        const int m = bm * BM + wm * 64 + i * 16 + rhalf + r;
        float v = acc[i][j][r];
        if constexpr (RESID) v += resid[(size_t)m * N + n];
        if constexpr (RELU) v = fmaxf(v, 0.0f);
        if constexpr (OUT_BF16)
          ((unsigned short*)Oop)[(size_t)m * N + n] = f2bf(v);
        else
          ((float*)Oop)[(size_t)m * N + n] = v;
      }
    }
}

// =============================================================================
// Flash attention: grid = (S/128, B*H). 256 thr = 8 waves, wave owns 16 rows.
// Q pre-scaled by 1/sqrt(dk). Q/K staged via GLOBAL_LOAD_ASYNC_TO_LDS_B128
// (overlapped with synchronous V transpose-staging), P fragment-linear.
// =============================================================================
__global__ __launch_bounds__(256) void attention_kernel(
    const unsigned short* __restrict__ Q, const unsigned short* __restrict__ K,
    const unsigned short* __restrict__ V, unsigned short* __restrict__ ctx) {
  constexpr int SEQ = 2048, D = 1024, DK = 64, QB = 128, KB = 64;
  constexpr int QS = DK + 8;   // 72 : sQ [m][d]
  constexpr int KS = DK + 8;   // 72 : sK [key][d]   (N-major for scores)
  constexpr int VT = KB + 8;   // 72 : sVT [d][key]  (N-major for PV)
  __shared__ unsigned short sQ[QB * QS];
  __shared__ unsigned short sK[KB * KS];
  __shared__ unsigned short sVT[DK * VT];
  __shared__ unsigned short sP[8 * 2 * 512];  // wave-private, fragment-linear

  const int tid = threadIdx.x, lane = tid & 31, wave = tid >> 5;
  const int qb = blockIdx.x;
  const int b = blockIdx.y >> 4, h = blockIdx.y & 15;
  const size_t base = (size_t)b * SEQ * D + (size_t)h * DK;
  const int nloc  = lane & 15;
  const int rhalf = (lane >> 4) << 3;

  // async-load Q block (pre-scaled): 1024 x b128
#pragma unroll
  for (int i = 0; i < 4; ++i) {
    const int idx = tid + i * 256;
    const int row = idx >> 3, c8 = idx & 7;
    async_b128(&sQ[row * QS + c8 * 8],
               Q + base + (size_t)(qb * QB + row) * D + c8 * 8);
  }

  float m_run[8], l_run[8];
#pragma unroll
  for (int r = 0; r < 8; ++r) { m_run[r] = -1e30f; l_run[r] = 0.0f; }
  v8f accO[4];
#pragma unroll
  for (int d = 0; d < 4; ++d)
#pragma unroll
    for (int r = 0; r < 8; ++r) accO[d][r] = 0.0f;

  unsigned short* myP = sP + wave * 1024;

  for (int kb = 0; kb < SEQ / KB; ++kb) {
    __syncthreads();  // previous iteration done reading sK/sVT
    // K block: async vector copy into natural [key][d] layout (512 x b128)
#pragma unroll
    for (int i = 0; i < 2; ++i) {
      const int idx = tid + i * 256;
      const int row = idx >> 3, c8 = idx & 7;
      async_b128(&sK[row * KS + c8 * 8],
                 K + base + (size_t)(kb * KB + row) * D + c8 * 8);
    }
    // V block: transpose-stage to [d][key]; two keys per thread so the
    // scattered stores are packed b32 along key (contiguous in [d][key]).
    {
      const int k0 = (tid >> 3) * 2;  // even key
      const int c8 = tid & 7;
      const uint4 v0 = *(const uint4*)(V + base + (size_t)(kb * KB + k0) * D + c8 * 8);
      const uint4 v1 = *(const uint4*)(V + base + (size_t)(kb * KB + k0 + 1) * D + c8 * 8);
      const int d0 = c8 * 8;
      const unsigned lo[4] = {v0.x, v0.y, v0.z, v0.w};
      const unsigned hi[4] = {v1.x, v1.y, v1.z, v1.w};
#pragma unroll
      for (int t = 0; t < 4; ++t) {
        *(unsigned*)&sVT[(d0 + 2 * t + 0) * VT + k0] = (lo[t] & 0xFFFFu) | (hi[t] << 16);
        *(unsigned*)&sVT[(d0 + 2 * t + 1) * VT + k0] = (lo[t] >> 16) | (hi[t] & 0xFFFF0000u);
      }
    }
    wait_async0();
    __syncthreads();

    // S = Q @ K^T (16 x 64 per wave): 8 wmma
    v8f s[4];
#pragma unroll
    for (int n = 0; n < 4; ++n)
#pragma unroll
      for (int r = 0; r < 8; ++r) s[n][r] = 0.0f;
#pragma unroll
    for (int ks = 0; ks < 2; ++ks) {
      const v16bf a = load_frag_a(&sQ[(wave * 16) * QS + ks * 32], QS);
#pragma unroll
      for (int n = 0; n < 4; ++n) {
        const v16bf bfr = load_frag_b_nm(&sK[(n * 16) * KS + ks * 32], KS);
        s[n] = wmma_bf16(a, bfr, s[n]);
      }
    }

    // online softmax (row's 16 values live in one 16-lane half per VGPR idx)
#pragma unroll
    for (int r = 0; r < 8; ++r) {
      float mx = s[0][r];
#pragma unroll
      for (int n = 1; n < 4; ++n) mx = fmaxf(mx, s[n][r]);
#pragma unroll
      for (int off = 1; off < 16; off <<= 1) mx = fmaxf(mx, __shfl_xor(mx, off, 32));
      const float mnew  = fmaxf(m_run[r], mx);
      const float alpha = __expf(m_run[r] - mnew);
#pragma unroll
      for (int d = 0; d < 4; ++d) accO[d][r] *= alpha;
      float rowsum = 0.0f;
#pragma unroll
      for (int n = 0; n < 4; ++n) {
        const float p = __expf(s[n][r] - mnew);
        s[n][r] = p;
        rowsum += p;
      }
#pragma unroll
      for (int off = 1; off < 16; off <<= 1) rowsum += __shfl_xor(rowsum, off, 32);
      l_run[r] = l_run[r] * alpha + rowsum;
      m_run[r] = mnew;
    }

    // scatter P into wave-private FRAGMENT-LINEAR layout:
    // element (m, k): owner lane = m + 16*((k>>3)&1); e = (k&7) + 8*((k>>4)&1)
#pragma unroll
    for (int n = 0; n < 4; ++n) {
      const int ks = n >> 1;
      const int kk = (n & 1) * 16 + nloc;
      const int e  = (kk & 7) + ((kk >> 4) & 1) * 8;
      const int oh = ((kk >> 3) & 1) * 16;
#pragma unroll
      for (int r = 0; r < 8; ++r) {
        const int m = rhalf + r;
        myP[ks * 512 + (m + oh) * 16 + e] = f2bf(s[n][r]);
      }
    }
    __syncthreads();

    // O += P @ V : 8 wmma
#pragma unroll
    for (int ks = 0; ks < 2; ++ks) {
      const v16bf a = load_frag_lin(&myP[ks * 512]);
#pragma unroll
      for (int d = 0; d < 4; ++d) {
        const v16bf bfr = load_frag_b_nm(&sVT[(d * 16) * VT + ks * 32], VT);
        accO[d] = wmma_bf16(a, bfr, accO[d]);
      }
    }
  }

  // epilogue: ctx[b, row, h*64+d] = O / l
#pragma unroll
  for (int d = 0; d < 4; ++d)
#pragma unroll
    for (int r = 0; r < 8; ++r) {
      const float o = accO[d][r] / l_run[r];
      const int m = qb * QB + wave * 16 + rhalf + r;
      ctx[base + (size_t)m * D + d * 16 + nloc] = f2bf(o);
    }
}

// =============================================================================
// Row LayerNorm over D=1024 (fp32), optional bf16 mirror.
// =============================================================================
__global__ __launch_bounds__(256) void layernorm_kernel(
    const float* __restrict__ in, const float* __restrict__ gamma,
    const float* __restrict__ beta, float* __restrict__ outf,
    unsigned short* __restrict__ outb) {
  __shared__ float red[256];
  const int row = blockIdx.x, tid = threadIdx.x;
  const float4 v = ((const float4*)(in + (size_t)row * 1024))[tid];
  red[tid] = v.x + v.y + v.z + v.w;
  __syncthreads();
  for (int o = 128; o > 0; o >>= 1) {
    if (tid < o) red[tid] += red[tid + o];
    __syncthreads();
  }
  const float mu = red[0] * (1.0f / 1024.0f);
  __syncthreads();
  const float d0 = v.x - mu, d1 = v.y - mu, d2 = v.z - mu, d3 = v.w - mu;
  red[tid] = d0 * d0 + d1 * d1 + d2 * d2 + d3 * d3;
  __syncthreads();
  for (int o = 128; o > 0; o >>= 1) {
    if (tid < o) red[tid] += red[tid + o];
    __syncthreads();
  }
  const float rs = rsqrtf(red[0] * (1.0f / 1024.0f) + 1e-5f);
  const float4 g = ((const float4*)gamma)[tid];
  const float4 be = ((const float4*)beta)[tid];
  float4 y;
  y.x = d0 * rs * g.x + be.x;
  y.y = d1 * rs * g.y + be.y;
  y.z = d2 * rs * g.z + be.z;
  y.w = d3 * rs * g.w + be.w;
  ((float4*)(outf + (size_t)row * 1024))[tid] = y;
  if (outb) {
    uint2 pk;
    pk.x = pack2bf(y.x, y.y);
    pk.y = pack2bf(y.z, y.w);
    ((uint2*)(outb + (size_t)row * 1024))[tid] = pk;
  }
}

// =============================================================================
extern "C" void kernel_launch(void* const* d_in, const int* in_sizes, int n_in,
                              void* d_out, int out_size, void* d_ws, size_t ws_size,
                              hipStream_t stream) {
  (void)in_sizes; (void)n_in; (void)out_size; (void)ws_size;
  const float* x   = (const float*)d_in[0];
  const float* Wq  = (const float*)d_in[1];
  const float* bq  = (const float*)d_in[2];
  const float* Wk  = (const float*)d_in[3];
  const float* bk  = (const float*)d_in[4];
  const float* Wv  = (const float*)d_in[5];
  const float* bv  = (const float*)d_in[6];
  const float* Wo  = (const float*)d_in[7];
  const float* bo  = (const float*)d_in[8];
  const float* W1  = (const float*)d_in[9];
  const float* b1  = (const float*)d_in[10];
  const float* W2  = (const float*)d_in[11];
  const float* b2  = (const float*)d_in[12];
  const float* g1  = (const float*)d_in[13];
  const float* be1 = (const float*)d_in[14];
  const float* g2  = (const float*)d_in[15];
  const float* be2 = (const float*)d_in[16];

  constexpr int ROWS = 2 * 2048;
  constexpr int Dm = 1024, Dff = 4096;

  char* w = (char*)d_ws;
  auto carve = [&](size_t bytes) {
    char* p = w;
    w += (bytes + 255) & ~(size_t)255;
    return p;
  };
  unsigned short* Qb   = (unsigned short*)carve((size_t)ROWS * Dm * 2);
  unsigned short* Kb   = (unsigned short*)carve((size_t)ROWS * Dm * 2);
  unsigned short* Vb   = (unsigned short*)carve((size_t)ROWS * Dm * 2);
  unsigned short* ctx  = (unsigned short*)carve((size_t)ROWS * Dm * 2);
  float*          tmp  = (float*)carve((size_t)ROWS * Dm * 4);
  float*          rm_f = (float*)carve((size_t)ROWS * Dm * 4);
  unsigned short* rm_b = (unsigned short*)carve((size_t)ROWS * Dm * 2);
  unsigned short* h1   = (unsigned short*)carve((size_t)ROWS * Dff * 2);

  const dim3 blk(256);
  const dim3 g_proj(Dm / 128, ROWS / 128);
  const dim3 g_ff1(Dff / 128, ROWS / 128);

  // QKV projections; 1/sqrt(dk) folded into the Q projection
  gemm_kernel<false, true, false, false><<<g_proj, blk, 0, stream>>>(x, Wq, bq, nullptr, Qb, ROWS, Dm, Dm, 0.125f);
  gemm_kernel<false, true, false, false><<<g_proj, blk, 0, stream>>>(x, Wk, bk, nullptr, Kb, ROWS, Dm, Dm, 1.0f);
  gemm_kernel<false, true, false, false><<<g_proj, blk, 0, stream>>>(x, Wv, bv, nullptr, Vb, ROWS, Dm, Dm, 1.0f);

  // Flash attention
  attention_kernel<<<dim3(2048 / 128, 32), blk, 0, stream>>>(Qb, Kb, Vb, ctx);

  // Output projection + residual(x); LN1 (fp32 + bf16 mirror)
  gemm_kernel<true, false, false, true><<<g_proj, blk, 0, stream>>>(ctx, Wo, bo, x, tmp, ROWS, Dm, Dm, 1.0f);
  layernorm_kernel<<<ROWS, blk, 0, stream>>>(tmp, g1, be1, rm_f, rm_b);

  // FFN + LN2 -> d_out
  gemm_kernel<true, true, true, false><<<g_ff1, blk, 0, stream>>>(rm_b, W1, b1, nullptr, h1, ROWS, Dff, Dm, 1.0f);
  gemm_kernel<true, false, false, true><<<g_proj, blk, 0, stream>>>(h1, W2, b2, rm_f, tmp, ROWS, Dm, Dff, 1.0f);
  layernorm_kernel<<<ROWS, blk, 0, stream>>>(tmp, g2, be2, (float*)d_out, nullptr);
}
